// MLPModel_32813550141718
// MI455X (gfx1250) — compile-verified, matrix operand order checked by reading, not run on hardware
//
#include <hip/hip_runtime.h>
#include <hip/hip_bf16.h>

typedef __attribute__((ext_vector_type(16))) _Float16 v16h;
typedef __attribute__((ext_vector_type(8)))  _Float16 v8h;
typedef __attribute__((ext_vector_type(8)))  float    v8f;

#define WMMA_F16(A, B, C) \
    __builtin_amdgcn_wmma_f32_16x16x32_f16(false, (A), false, (B), (short)0, (C), false, false)

// ---- hash grid config (matches reference) ----
static constexpr int kLevels = 16;
static constexpr unsigned kT = 1u << 19;
static constexpr int kRes[16] = {16, 24, 36, 54, 81, 121, 182, 273,
                                 410, 615, 922, 1383, 2075, 3113, 4670, 7006};
// levels 0..3 are dense ((res+1)^3 <= T), 4..15 hashed

__device__ __forceinline__ v16h cat8(v8h lo, v8h hi) {
    return __builtin_shufflevector(lo, hi, 0, 1, 2, 3, 4, 5, 6, 7,
                                   8, 9, 10, 11, 12, 13, 14, 15);
}

// Load WMMA A operand (16-bit, 16x32 tile) for this lane from a row-major
// f16 activation row. VGPR0-3 hold K=kb..kb+7, VGPR4-7 hold K=kb+16..kb+23.
__device__ __forceinline__ v16h load_a(const _Float16* rowptr, int kb) {
    v8h lo = *(const v8h*)(rowptr + kb);
    v8h hi = *(const v8h*)(rowptr + kb + 16);
    return cat8(lo, hi);
}

// Pre-swizzle a KxN f32 weight matrix into the per-lane WMMA B layout:
// dst[((j*kchunks + c)*32 + lane)*16 + i]  holds the v16h lane 'lane' needs
// for N-tile j, K-chunk c.
__device__ __forceinline__ void prep_weights(const float* __restrict__ W,
                                             _Float16* dst, int ncols,
                                             int kchunks, int ntiles) {
    const int total = ntiles * kchunks * 512;
    for (int s = threadIdx.x; s < total; s += 256) {
        int i = s & 15;
        int L = (s >> 4) & 31;
        int c = (s >> 9) % kchunks;
        int j = s / (512 * kchunks);
        int n  = j * 16 + (L & 15);
        int kb = c * 32 + ((L >> 4) << 3);
        int k  = kb + (i < 8 ? i : i + 8);
        dst[s] = (_Float16)W[k * ncols + n];
    }
}

// Encode 8 consecutive levels [L0, L0+8) of one point into 16 f16 features.
template <int L0>
__device__ __forceinline__ void encode8(float xx, float yy, float zz,
                                        const float* __restrict__ table,
                                        _Float16* dst) {
#pragma unroll
    for (int t = 0; t < 8; ++t) {
        const int lvl = L0 + t;
        const float resf = (float)kRes[lvl];
        float px = xx * resf, py = yy * resf, pz = zz * resf;
        float fx = floorf(px), fy = floorf(py), fz = floorf(pz);
        float wx = px - fx, wy = py - fy, wz = pz - fz;
        unsigned cx = (unsigned)fx, cy = (unsigned)fy, cz = (unsigned)fz;
        const float* tb = table + ((size_t)lvl << 20);  // lvl * T * 2
        float f0 = 0.f, f1 = 0.f;
#pragma unroll
        for (int corner = 0; corner < 8; ++corner) {
            unsigned oi = (corner >> 2) & 1u;
            unsigned oj = (corner >> 1) & 1u;
            unsigned ok = corner & 1u;
            unsigned ax = cx + oi, ay = cy + oj, az = cz + ok;
            unsigned idx;
            if (lvl < 4) {  // dense (folds at compile time after unroll)
                unsigned s = (unsigned)kRes[lvl] + 1u;
                idx = ax + ay * s + az * s * s;
            } else {        // tcnn spatial hash
                idx = (ax * 1u) ^ (ay * 2654435761u) ^ (az * 805459861u);
                idx &= (kT - 1u);
            }
            float2 f = *(const float2*)(tb + ((size_t)idx << 1));
            float wt = (oi ? wx : 1.f - wx) *
                       (oj ? wy : 1.f - wy) *
                       (ok ? wz : 1.f - wz);
            f0 = fmaf(wt, f.x, f0);
            f1 = fmaf(wt, f.y, f1);
        }
        dst[2 * t]     = (_Float16)f0;
        dst[2 * t + 1] = (_Float16)f1;
    }
}

// Fully-fused hashgrid + 32->64->64->16 MLP. 256 threads = 8 waves,
// 128 points per block, one 16-row WMMA M-tile per wave.
__global__ __launch_bounds__(256) void ngp_fused_kernel(
    const float* __restrict__ x, const float* __restrict__ table,
    const float* __restrict__ W1, const float* __restrict__ W2,
    const float* __restrict__ W3, float* __restrict__ out) {
    __shared__ __align__(32) _Float16 sFeat[128][32];      // 8 KB
    __shared__ __align__(32) _Float16 sH1[128][64];        // 16 KB
    __shared__ __align__(32) _Float16 sH2[128][64];        // 16 KB
    __shared__ __align__(32) _Float16 sB1[4 * 1 * 32 * 16];  // 4 KB
    __shared__ __align__(32) _Float16 sB2[4 * 2 * 32 * 16];  // 8 KB
    __shared__ __align__(32) _Float16 sB3[1 * 2 * 32 * 16];  // 2 KB

    const int tid = threadIdx.x;

    // ---- stage weights into WMMA-B-swizzled LDS (L2-resident reload) ----
    prep_weights(W1, sB1, 64, 1, 4);
    prep_weights(W2, sB2, 64, 2, 4);
    prep_weights(W3, sB3, 16, 2, 1);

    // ---- encode: 2 threads per point, 8 levels each ----
    const int p = tid >> 1;
    const int half = tid & 1;
    const long gp = (long)blockIdx.x * 128 + p;
    float xx = x[gp * 3 + 0];
    float yy = x[gp * 3 + 1];
    float zz = x[gp * 3 + 2];
    if (half == 0)
        encode8<0>(xx, yy, zz, table, &sFeat[p][0]);
    else
        encode8<8>(xx, yy, zz, table, &sFeat[p][16]);

    __syncthreads();

    // ---- fused MLP on WMMA ----
    const int lane = tid & 31;
    const int wid = tid >> 5;
    const int row = wid * 16 + (lane & 15);  // this lane's A-matrix row
    const int kb0 = (lane >> 4) << 3;        // K sub-offset per A layout
    const int mrow = ((lane >> 4) << 3);     // D-layout row offset
    const int ncol = lane & 15;              // D-layout column

    // Layer 1: [16x32] x [32x64], K=32 => 1 WMMA per N-tile
    v16h a1 = load_a(&sFeat[row][0], kb0);
    v8f h[4];
#pragma unroll
    for (int j = 0; j < 4; ++j) {
        v16h b = *(const v16h*)(sB1 + (j * 32 + lane) * 16);
        v8f c = {};
        h[j] = WMMA_F16(a1, b, c);
    }
#pragma unroll
    for (int j = 0; j < 4; ++j)
#pragma unroll
        for (int r = 0; r < 8; ++r)
            sH1[wid * 16 + r + mrow][j * 16 + ncol] =
                (_Float16)fmaxf(h[j][r], 0.f);  // fused ReLU

    __syncthreads();

    // Layer 2: [16x64] x [64x64], K=64 => 2 WMMAs per N-tile
    v16h a2lo = load_a(&sH1[row][0], kb0);
    v16h a2hi = load_a(&sH1[row][32], kb0);
    v8f g[4];
#pragma unroll
    for (int j = 0; j < 4; ++j) {
        v8f c = {};
        v16h b0 = *(const v16h*)(sB2 + ((j * 2 + 0) * 32 + lane) * 16);
        c = WMMA_F16(a2lo, b0, c);
        v16h b1 = *(const v16h*)(sB2 + ((j * 2 + 1) * 32 + lane) * 16);
        c = WMMA_F16(a2hi, b1, c);
        g[j] = c;
    }
#pragma unroll
    for (int j = 0; j < 4; ++j)
#pragma unroll
        for (int r = 0; r < 8; ++r)
            sH2[wid * 16 + r + mrow][j * 16 + ncol] =
                (_Float16)fmaxf(g[j][r], 0.f);  // fused ReLU

    __syncthreads();

    // Layer 3: [16x64] x [64x16], K=64 => 2 WMMAs, one N-tile
    v16h a3lo = load_a(&sH2[row][0], kb0);
    v16h a3hi = load_a(&sH2[row][32], kb0);
    v8f o = {};
    o = WMMA_F16(a3lo, *(const v16h*)(sB3 + (0 * 32 + lane) * 16), o);
    o = WMMA_F16(a3hi, *(const v16h*)(sB3 + (1 * 32 + lane) * 16), o);

    // D tile straight to global: out[point][0..15], f32
    const long obase = (long)blockIdx.x * 128 + wid * 16;
#pragma unroll
    for (int r = 0; r < 8; ++r)
        out[(obase + r + mrow) * 16 + ncol] = o[r];
}

extern "C" void kernel_launch(void* const* d_in, const int* in_sizes, int n_in,
                              void* d_out, int out_size, void* d_ws, size_t ws_size,
                              hipStream_t stream) {
    const float* x     = (const float*)d_in[0];
    const float* table = (const float*)d_in[1];
    const float* W1    = (const float*)d_in[2];
    const float* W2    = (const float*)d_in[3];
    const float* W3    = (const float*)d_in[4];
    float* out = (float*)d_out;

    const int npts = in_sizes[0] / 3;     // 2,097,152
    const int blocks = npts / 128;        // 128 points per 256-thread block

    ngp_fused_kernel<<<blocks, 256, 0, stream>>>(x, table, W1, W2, W3, out);
}
